// MultiHeadedAttention_9139690406316
// MI455X (gfx1250) — matrix operand order repaired and checked by reading, not compiled, not run on hardware
//
#include <hip/hip_runtime.h>
#include <hip/hip_bf16.h>

// ---------------- problem constants ----------------
constexpr int Bc  = 2;
constexpr int Sc  = 2048;
constexpr int Dc  = 1024;
constexpr int Hc  = 16;
constexpr int DKc = 64;

typedef _Float16 v16h __attribute__((ext_vector_type(16)));
typedef _Float16 v8h  __attribute__((ext_vector_type(8)));
typedef float    v8f  __attribute__((ext_vector_type(8)));
typedef int      v4i_t __attribute__((vector_size(16)));

// ---------------- fragment chunk loaders ----------------
__device__ __forceinline__ void ld8(v16h& d, int o, const _Float16* p) {
  v8h t = *(const v8h*)p;                       // 16B vector load
#pragma unroll
  for (int i = 0; i < 8; ++i) d[o + i] = t[i];
}
__device__ __forceinline__ void ld8(v16h& d, int o, const float* p) {
  float4 t0 = *(const float4*)p;
  float4 t1 = *(const float4*)(p + 4);
  d[o + 0] = (_Float16)t0.x; d[o + 1] = (_Float16)t0.y;
  d[o + 2] = (_Float16)t0.z; d[o + 3] = (_Float16)t0.w;
  d[o + 4] = (_Float16)t1.x; d[o + 5] = (_Float16)t1.y;
  d[o + 6] = (_Float16)t1.z; d[o + 7] = (_Float16)t1.w;
}

__device__ __forceinline__ v8f wmma_f16(const v16h& a, const v16h& b, const v8f& c) {
  return __builtin_amdgcn_wmma_f32_16x16x32_f16(
      /*neg_a=*/false, a, /*neg_b=*/false, b,
      /*c_mod=*/(short)0, c, /*reuse_a=*/false, /*reuse_b=*/false);
}

// ---------------- async global -> LDS (16B per lane, ASYNCcnt-tracked) ----------
#if defined(__has_builtin)
#if __has_builtin(__builtin_amdgcn_global_load_async_to_lds_b128)
#define HAVE_ASYNC_LDS_BUILTIN 1
#endif
#endif
#ifndef HAVE_ASYNC_LDS_BUILTIN
#define HAVE_ASYNC_LDS_BUILTIN 0
#endif

__device__ __forceinline__ void async_copy16(void* lds_dst, const void* gsrc) {
#if HAVE_ASYNC_LDS_BUILTIN
  __builtin_amdgcn_global_load_async_to_lds_b128(
      (__attribute__((address_space(1))) v4i_t*)gsrc,
      (__attribute__((address_space(3))) v4i_t*)lds_dst,
      /*offset=*/0, /*cpol=*/0);
#else
  // GV mode: vdst = LDS byte address, vaddr = 64-bit global address.
  unsigned lofs = (unsigned)(size_t)lds_dst;     // low 32 bits == LDS offset
  asm volatile("global_load_async_to_lds_b128 %0, %1, off"
               :: "v"(lofs), "v"((unsigned long long)(size_t)gsrc)
               : "memory");
#endif
}

// ---------------- GEMM: out[m,n] = sum_k A[m,k]*W[n,k] + bias[n] ----------------
// Store modes: 0 = f16 [B,H,S,DK] (Q/K, scale folded), 1 = f16 [B,H,DK,S] (V^T),
//              2 = f32 row-major [M,N] (final output)
template <typename AT, int SM>
__global__ __launch_bounds__(256) void gemm_wmma_kernel(
    const AT* __restrict__ A, const float* __restrict__ W,
    const float* __restrict__ bias, void* __restrict__ out,
    int M, int N, int K, float scale) {
  const int lane = threadIdx.x & 31;
  const int wave = threadIdx.x >> 5;
  const int half = lane >> 4;
  const int r    = lane & 15;
  const int wm   = wave & 3;                 // 4 waves along M
  const int wn   = wave >> 2;                // 2 waves along N
  const int m0   = blockIdx.y * 128 + wm * 32;
  const int n0   = blockIdx.x * 128 + wn * 64;

  v8f acc[2][4];
#pragma unroll
  for (int mi = 0; mi < 2; ++mi)
#pragma unroll
    for (int ni = 0; ni < 4; ++ni) acc[mi][ni] = (v8f)0.0f;

  for (int k0 = 0; k0 < K; k0 += 32) {
    // A fragments: row = m, K chunks {half*8 .. +7} and {half*8+16 .. +7}
    v16h a[2];
#pragma unroll
    for (int mi = 0; mi < 2; ++mi) {
      const AT* ap = A + (size_t)(m0 + mi * 16 + r) * K + k0 + half * 8;
      ld8(a[mi], 0, ap);
      ld8(a[mi], 8, ap + 16);
      if (k0 + 32 < K) __builtin_prefetch(ap + 32, 0, 0);
    }
    // B fragments from W (row n holds its K values contiguously)
    v16h b[4];
#pragma unroll
    for (int ni = 0; ni < 4; ++ni) {
      const float* wp = W + (size_t)(n0 + ni * 16 + r) * K + k0 + half * 8;
      ld8(b[ni], 0, wp);
      ld8(b[ni], 8, wp + 16);
    }
#pragma unroll
    for (int mi = 0; mi < 2; ++mi)
#pragma unroll
      for (int ni = 0; ni < 4; ++ni)
        acc[mi][ni] = wmma_f16(a[mi], b[ni], acc[mi][ni]);
  }

  // epilogue: C layout -> element (m = base + j + half*8, n = base + r)
#pragma unroll
  for (int ni = 0; ni < 4; ++ni) {
    const int n  = n0 + ni * 16 + r;
    const float bn = bias[n];
#pragma unroll
    for (int mi = 0; mi < 2; ++mi) {
#pragma unroll
      for (int j = 0; j < 8; ++j) {
        const int m = m0 + mi * 16 + j + half * 8;
        const float v = (acc[mi][ni][j] + bn) * scale;
        if constexpr (SM == 2) {
          ((float*)out)[(size_t)m * N + n] = v;
        } else {
          const int bb = m / Sc, s = m % Sc;
          const int h = n / DKc, dk = n % DKc;
          if constexpr (SM == 0)
            ((_Float16*)out)[(((size_t)bb * Hc + h) * Sc + s) * DKc + dk] = (_Float16)v;
          else
            ((_Float16*)out)[(((size_t)bb * Hc + h) * DKc + dk) * Sc + s] = (_Float16)v;
        }
      }
    }
  }
}

// ---------------- flash attention (async double-buffered K/V tiles) ----------
// grid: (S/128, B*H); block 256 (8 waves). Wave w owns 16 query rows.
__global__ __launch_bounds__(256) void attn_flash_kernel(
    const _Float16* __restrict__ Q,   // [B,H,S,DK], pre-scaled by 1/sqrt(DK)
    const _Float16* __restrict__ Kf,  // [B,H,S,DK]
    const _Float16* __restrict__ Vt,  // [B,H,DK,S]
    const int* __restrict__ mask,     // [B,S,S]
    _Float16* __restrict__ Cx) {      // [B,S,D] f16
  __shared__ _Float16 Kl[2][32 * 64];     // double-buffered kv-block x dk
  __shared__ _Float16 Vl[2][64 * 32];     // double-buffered dk x kv-block
  __shared__ _Float16 Pl[8 * 16 * 32];    // per-wave P relayout scratch

  const int tid  = threadIdx.x;
  const int lane = tid & 31;
  const int wave = tid >> 5;
  const int half = lane >> 4;
  const int r    = lane & 15;
  const int bh   = blockIdx.y;
  const int b    = bh >> 4;               // / H
  const int h    = bh & 15;
  const int q0   = blockIdx.x * 128 + wave * 16;

  // per-thread coop-load coordinates (16B chunk each for K and V tiles)
  const int kvl = tid >> 3, dc = (tid & 7) * 8;   // K tile: 32 x 64
  const int dkl = tid >> 2, kc = (tid & 3) * 8;   // V tile: 64 x 32
  const _Float16* Kg = Kf + (size_t)bh * Sc * DKc;
  const _Float16* Vg = Vt + (size_t)bh * DKc * Sc;

  // Q A-fragments (kept in registers for the whole kv sweep)
  v16h aq[2];
  {
    const _Float16* qp = Q + ((size_t)bh * Sc + (q0 + r)) * DKc;
#pragma unroll
    for (int ks = 0; ks < 2; ++ks) {
      ld8(aq[ks], 0, qp + ks * 32 + half * 8);
      ld8(aq[ks], 8, qp + ks * 32 + half * 8 + 16);
    }
  }

  v8f acc[4];
#pragma unroll
  for (int dt = 0; dt < 4; ++dt) acc[dt] = (v8f)0.0f;
  float mrow[8], lrow[8];
#pragma unroll
  for (int j = 0; j < 8; ++j) { mrow[j] = -1.0e30f; lrow[j] = 0.0f; }

  _Float16* Plw = Pl + wave * 512;

  // prologue: issue async loads for stage 0 (2 ops/lane -> ASYNCcnt = 2)
  async_copy16(&Kl[0][kvl * 64 + dc], &Kg[(size_t)(0 + kvl) * DKc + dc]);
  async_copy16(&Vl[0][dkl * 32 + kc], &Vg[(size_t)dkl * Sc + 0 + kc]);

  const int nIter = Sc / 32;
  for (int it = 0; it < nIter; ++it) {
    const int kv0 = it * 32;
    const _Float16* Kc = Kl[it & 1];
    const _Float16* Vc = Vl[it & 1];

    __syncthreads();   // everyone done reading the buffer we overwrite next
    if (it + 1 < nIter) {
      const int kvn = kv0 + 32;
      _Float16* Kn = Kl[(it + 1) & 1];
      _Float16* Vn = Vl[(it + 1) & 1];
      async_copy16(&Kn[kvl * 64 + dc], &Kg[(size_t)(kvn + kvl) * DKc + dc]);
      async_copy16(&Vn[dkl * 32 + kc], &Vg[(size_t)dkl * Sc + kvn + kc]);
      // 4 outstanding; wait until only the 2 just-issued remain (in-order)
      asm volatile("s_wait_asynccnt 0x2" ::: "memory");
    } else {
      asm volatile("s_wait_asynccnt 0x0" ::: "memory");
    }
    __syncthreads();   // all waves' chunks of the current tile have landed

    // ---- scores S = Q K^T (already includes 1/sqrt(DK) via Q) ----
    float ps[2][8];
#pragma unroll
    for (int nt = 0; nt < 2; ++nt) {
      v8f c = (v8f)0.0f;
#pragma unroll
      for (int ks = 0; ks < 2; ++ks) {
        v16h bk;
        const _Float16* kp = &Kc[(nt * 16 + r) * 64 + ks * 32 + half * 8];
        ld8(bk, 0, kp);
        ld8(bk, 8, kp + 16);
        c = wmma_f16(aq[ks], bk, c);
      }
#pragma unroll
      for (int j = 0; j < 8; ++j) {
        const int q    = q0 + j + half * 8;
        const int kcol = kv0 + nt * 16 + r;
        const int mv   = mask[((size_t)b * Sc + q) * Sc + kcol];
        ps[nt][j] = (mv == 0) ? -1.0e9f : c[j];
      }
    }

    // ---- online softmax update ----
#pragma unroll
    for (int j = 0; j < 8; ++j) {
      float t = fmaxf(ps[0][j], ps[1][j]);
#pragma unroll
      for (int off = 1; off <= 8; off <<= 1)
        t = fmaxf(t, __shfl_xor(t, off, 32));      // reduce within 16-lane group
      const float mn = fmaxf(mrow[j], t);
      const float f  = __expf(mrow[j] - mn);
      mrow[j] = mn;
      lrow[j] *= f;
#pragma unroll
      for (int dt = 0; dt < 4; ++dt) acc[dt][j] *= f;
      float p0 = __expf(ps[0][j] - mn);
      float p1 = __expf(ps[1][j] - mn);
      ps[0][j] = p0; ps[1][j] = p1;
      float rs = p0 + p1;
#pragma unroll
      for (int off = 1; off <= 8; off <<= 1)
        rs += __shfl_xor(rs, off, 32);
      lrow[j] += rs;
    }

    // ---- relayout P (C-frag -> A-frag) via per-wave LDS tile ----
#pragma unroll
    for (int nt = 0; nt < 2; ++nt)
#pragma unroll
      for (int j = 0; j < 8; ++j)
        Plw[(j + half * 8) * 32 + nt * 16 + r] = (_Float16)ps[nt][j];
    asm volatile("s_wait_dscnt 0" ::: "memory");   // per-wave LDS RAW fence

    v16h ap;
    ld8(ap, 0, &Plw[r * 32 + half * 8]);
    ld8(ap, 8, &Plw[r * 32 + half * 8 + 16]);

    // ---- ctx += P @ V ----
#pragma unroll
    for (int dt = 0; dt < 4; ++dt) {
      v16h bv;
      const _Float16* vp = &Vc[(dt * 16 + r) * 32 + half * 8];
      ld8(bv, 0, vp);
      ld8(bv, 8, vp + 16);
      acc[dt] = wmma_f16(ap, bv, acc[dt]);
    }
  }

  // ---- normalize and write ctx back as f16 [B,S,D] (head-interleaved) ----
#pragma unroll
  for (int dt = 0; dt < 4; ++dt) {
#pragma unroll
    for (int j = 0; j < 8; ++j) {
      const int q   = q0 + j + half * 8;
      const int col = h * DKc + dt * 16 + r;
      Cx[((size_t)b * Sc + q) * Dc + col] = (_Float16)(acc[dt][j] / lrow[j]);
    }
  }
}

// ---------------- host launcher ----------------
extern "C" void kernel_launch(void* const* d_in, const int* in_sizes, int n_in,
                              void* d_out, int out_size, void* d_ws, size_t ws_size,
                              hipStream_t stream) {
  const float* query = (const float*)d_in[0];
  const float* key   = (const float*)d_in[1];
  const float* value = (const float*)d_in[2];
  const int*   mask  = (const int*)d_in[3];
  const float* Wq = (const float*)d_in[4];
  const float* bq = (const float*)d_in[5];
  const float* Wk = (const float*)d_in[6];
  const float* bk = (const float*)d_in[7];
  const float* Wv = (const float*)d_in[8];
  const float* bv = (const float*)d_in[9];
  const float* Wo = (const float*)d_in[10];
  const float* bo = (const float*)d_in[11];
  float* out = (float*)d_out;

  const size_t elems = (size_t)Bc * Sc * Dc;   // 4M elements
  _Float16* Qf = (_Float16*)d_ws;
  _Float16* Kf = Qf + elems;
  _Float16* Vt = Kf + elems;
  _Float16* Cx = Vt + elems;

  const int M = Bc * Sc, N = Dc, K = Dc;
  dim3 gblk(256);
  dim3 ggrid(N / 128, M / 128);

  // Q projection, scale 1/sqrt(DK) folded in; layout [B,H,S,DK]
  gemm_wmma_kernel<float, 0><<<ggrid, gblk, 0, stream>>>(query, Wq, bq, Qf, M, N, K, 0.125f);
  // K projection; layout [B,H,S,DK]
  gemm_wmma_kernel<float, 0><<<ggrid, gblk, 0, stream>>>(key, Wk, bk, Kf, M, N, K, 1.0f);
  // V projection; transposed layout [B,H,DK,S]
  gemm_wmma_kernel<float, 1><<<ggrid, gblk, 0, stream>>>(value, Wv, bv, Vt, M, N, K, 1.0f);

  // flash attention
  dim3 agrid(Sc / 128, Bc * Hc);
  attn_flash_kernel<<<agrid, gblk, 0, stream>>>(Qf, Kf, Vt, mask, Cx);

  // output projection to fp32
  gemm_wmma_kernel<_Float16, 2><<<ggrid, gblk, 0, stream>>>(Cx, Wo, bo, out, M, N, K, 1.0f);
}
